// ContinuousDepthGeneModule_25185688224023
// MI455X (gfx1250) — matrix operand order repaired and checked by reading, not compiled, not run on hardware
//
#include <hip/hip_runtime.h>

#define NN 20000
#define EE 320000
#define DH 128
#define TP 20
#define EPS_LN 1e-5f

typedef unsigned int u32;
typedef unsigned short u16;
typedef __attribute__((ext_vector_type(16))) __bf16 v16bf;
typedef __attribute__((ext_vector_type(8)))  float  v8f;

union Frag { v16bf v; u32 u[8]; };

__device__ __forceinline__ u16 f2bf(float f) {
  u32 u = __float_as_uint(f);
  u32 r = (u + 0x7FFFu + ((u >> 16) & 1u)) >> 16;  // round-to-nearest-even
  return (u16)r;
}
__device__ __forceinline__ u32 f2bf2(float lo, float hi) {
  return (u32)f2bf(lo) | ((u32)f2bf(hi) << 16);
}
__device__ __forceinline__ float bf2f(u32 bits16) {
  return __uint_as_float(bits16 << 16);
}
__device__ __forceinline__ float sigm(float x) { return 1.f / (1.f + __expf(-x)); }

// ---------------- scalar prep: dt, epigenetic scale, zero d_out ----------------
__global__ void k_prep(const float* __restrict__ meth, const float* __restrict__ hist,
                       const float* __restrict__ logd, float* __restrict__ scal,
                       float* __restrict__ dout) {
  __shared__ float sm[128];
  int t = threadIdx.x;
  sm[t] = sigm(meth[t]);
  dout[t] = 0.f;
  __syncthreads();
  if (t == 0) {
    float s = 0.f;
    for (int i = 0; i < 128; ++i) s += sm[i];
    float msil = s * (1.f / 128.f);
    float h0 = sigm(hist[0]), h1 = sigm(hist[1]), h2 = sigm(hist[2]), h3 = sigm(hist[3]);
    float access = fminf(fmaxf(0.5f * (h0 + h2) - 0.5f * (h1 + h3) + 0.5f, 0.f), 1.f);
    scal[1] = access * (1.f - msil);
    float depth = fminf(fmaxf(__expf(logd[0]), 0.1f), 3.0f);
    scal[0] = depth / (float)(TP - 1);
  }
}

// ---------------- degree / normalization / CSR build ----------------
__global__ void k_deg(const int* __restrict__ ei, float* __restrict__ deg) {
  int e = blockIdx.x * 256 + threadIdx.x;
  if (e < EE) atomicAdd(&deg[ei[EE + e]], 1.0f);
}
__global__ void k_dis(const float* __restrict__ deg, float* __restrict__ dis) {
  int i = blockIdx.x * 256 + threadIdx.x;
  if (i < NN) {
    float d = deg[i];
    dis[i] = (d > 0.f) ? rsqrtf(fmaxf(d, 1.f)) : 0.f;
  }
}
__global__ void k_scan(const float* __restrict__ deg, int* __restrict__ off) {
  __shared__ int lds[1024];
  const int t = threadIdx.x;
  const int CH = 20;  // 1024*20 >= NN
  int s0 = t * CH, sum = 0;
  for (int i = 0; i < CH; ++i) {
    int idx = s0 + i;
    if (idx < NN) sum += (int)deg[idx];
  }
  lds[t] = sum;
  __syncthreads();
  if (t == 0) {
    int run = 0;
    for (int i = 0; i < 1024; ++i) { int v = lds[i]; lds[i] = run; run += v; }
  }
  __syncthreads();
  int base = lds[t];
  for (int i = 0; i < CH; ++i) {
    int idx = s0 + i;
    if (idx < NN) { off[idx] = base; base += (int)deg[idx]; }
  }
  if (t == 0) off[NN] = EE;
}
__global__ void k_fill(const int* __restrict__ ei, const int* __restrict__ off,
                       int* __restrict__ cur, int* __restrict__ srcA) {
  int e = blockIdx.x * 256 + threadIdx.x;
  if (e < EE) {
    int c = ei[EE + e];
    int pos = off[c] + atomicAdd(&cur[c], 1);
    srcA[pos] = ei[e];
  }
}

// ---------------- f32 -> bf16 conversions ----------------
__global__ void k_conv(const float* __restrict__ in, u16* __restrict__ out, int n) {
  int i = blockIdx.x * 256 + threadIdx.x;
  if (i < n) out[i] = f2bf(in[i]);
}
// in: row-major [K][Nn] f32, out: [Nn][K] bf16 (K-major, so WMMA B K-pairs are contiguous)
__global__ void k_convT(const float* __restrict__ in, u16* __restrict__ out, int K, int Nn) {
  int i = blockIdx.x * 256 + threadIdx.x;
  if (i < K * Nn) {
    int k = i / Nn, n = i - k * Nn;
    out[n * K + k] = f2bf(in[i]);
  }
}

// ---------------- WMMA GEMM: C[64rows/block,128] = A[.,K] * W[K,128] + fused epilogue
// MODE 0: input proj (+bias, LN, ReLU, *epi) -> outf f32 + outb bf16
// MODE 1: node GCN  (*dis[row])              -> outb bf16 (tmp for gather)
// MODE 2: gate      (sigmoid(+bias), blend hc/hn in place over hn) -> outf/outb
// MODE 3: output    (+bias, LN, mean-pool atomics into dout)
template <int PASSES, int PK, int MODE>
__global__ __launch_bounds__(128) void k_gemm(
    const u16* __restrict__ A1, const u16* __restrict__ A2, const u16* __restrict__ Wt,
    const float* __restrict__ bias, const float* __restrict__ lng, const float* __restrict__ lnb,
    const float* __restrict__ dis, const float* __restrict__ hcf,
    float* __restrict__ outf, u16* __restrict__ outb,
    const float* __restrict__ scal, float* __restrict__ dout) {
  __shared__ u32 wl[128 * 66];  // W tile: 128 n-rows x PK K-pairs (padded pitch 66)
  __shared__ u32 al[64 * 66];   // A tile: 64 rows x PK K-pairs

  const int tid = threadIdx.x;
  const int lane = tid & 31, wid = tid >> 5;
  const int lo = lane & 15, hi = lane >> 4;
  const int R0 = blockIdx.x * 64;
  const int PKTOT = PASSES * PK;

  v8f acc[8];
  const v8f vzero = {0.f, 0.f, 0.f, 0.f, 0.f, 0.f, 0.f, 0.f};
#pragma unroll
  for (int nt = 0; nt < 8; ++nt) acc[nt] = vzero;

  const u32* WtU = (const u32*)Wt;
  for (int p = 0; p < PASSES; ++p) {
    if (p) __syncthreads();
    for (int i = tid; i < 128 * PK; i += 128) {
      int n = i / PK, j = i - n * PK;
      wl[n * 66 + j] = WtU[n * PKTOT + p * PK + j];
    }
    const u32* AU = (const u32*)((p == 0) ? A1 : A2);
    for (int i = tid; i < 64 * PK; i += 128) {
      int r = i / PK, j = i - r * PK;
      int gr = R0 + r;
      al[r * 66 + j] = (gr < NN) ? AU[gr * PK + j] : 0u;
    }
    __syncthreads();
    const int KT = PK / 16;  // 16 K-pairs (=32 K) per WMMA
    for (int kt = 0; kt < KT; ++kt) {
      Frag a;
#pragma unroll
      for (int v = 0; v < 8; ++v)
        a.u[v] = al[(wid * 16 + lo) * 66 + kt * 16 + (v < 4 ? 0 : 8) + hi * 4 + (v & 3)];
#pragma unroll
      for (int nt = 0; nt < 8; ++nt) {
        Frag b;
#pragma unroll
        for (int v = 0; v < 8; ++v)
          b.u[v] = wl[(nt * 16 + lo) * 66 + kt * 16 + (v < 4 ? 0 : 8) + hi * 4 + (v & 3)];
        acc[nt] = __builtin_amdgcn_wmma_f32_16x16x32_bf16(
            false, a.v, false, b.v, (short)0, acc[nt], false, false);
      }
    }
  }
  __syncthreads();

  // ---- epilogue: stage C tiles in LDS (reuse wl), process row-wise ----
  float* et = ((float*)wl) + wid * (16 * 132);
#pragma unroll
  for (int nt = 0; nt < 8; ++nt)
#pragma unroll
    for (int r = 0; r < 8; ++r)
      et[(r + hi * 8) * 132 + nt * 16 + lo] = acc[nt][r];
  __syncthreads();

  float cs0 = 0.f, cs1 = 0.f, cs2 = 0.f, cs3 = 0.f;
  const int c = lane * 4;
  for (int m = 0; m < 16; ++m) {
    const int gr = R0 + wid * 16 + m;
    if (gr >= NN) break;
    const float* rowp = et + m * 132 + c;
    float v0 = rowp[0], v1 = rowp[1], v2 = rowp[2], v3 = rowp[3];
    if constexpr (MODE == 1) {
      const float dn = dis[gr];
      ((uint2*)outb)[gr * 32 + lane] =
          make_uint2(f2bf2(v0 * dn, v1 * dn), f2bf2(v2 * dn, v3 * dn));
    } else if constexpr (MODE == 2) {
      float g0 = sigm(v0 + bias[c + 0]), g1 = sigm(v1 + bias[c + 1]);
      float g2 = sigm(v2 + bias[c + 2]), g3 = sigm(v3 + bias[c + 3]);
      float* hn = outf + gr * DH + c;
      const float* hc = hcf + gr * DH + c;
      float o0 = g0 * hn[0] + (1.f - g0) * hc[0];
      float o1 = g1 * hn[1] + (1.f - g1) * hc[1];
      float o2 = g2 * hn[2] + (1.f - g2) * hc[2];
      float o3 = g3 * hn[3] + (1.f - g3) * hc[3];
      hn[0] = o0; hn[1] = o1; hn[2] = o2; hn[3] = o3;
      ((uint2*)outb)[gr * 32 + lane] = make_uint2(f2bf2(o0, o1), f2bf2(o2, o3));
    } else {
      v0 += bias[c + 0]; v1 += bias[c + 1]; v2 += bias[c + 2]; v3 += bias[c + 3];
      float s1 = v0 + v1 + v2 + v3;
      float s2 = v0 * v0 + v1 * v1 + v2 * v2 + v3 * v3;
#pragma unroll
      for (int o = 16; o >= 1; o >>= 1) {
        s1 += __shfl_xor(s1, o, 32);
        s2 += __shfl_xor(s2, o, 32);
      }
      const float mu = s1 * (1.f / 128.f);
      const float rs = rsqrtf(s2 * (1.f / 128.f) - mu * mu + EPS_LN);
      float y0 = (v0 - mu) * rs * lng[c + 0] + lnb[c + 0];
      float y1 = (v1 - mu) * rs * lng[c + 1] + lnb[c + 1];
      float y2 = (v2 - mu) * rs * lng[c + 2] + lnb[c + 2];
      float y3 = (v3 - mu) * rs * lng[c + 3] + lnb[c + 3];
      if constexpr (MODE == 0) {
        const float ep = scal[1];
        y0 = fmaxf(y0, 0.f) * ep; y1 = fmaxf(y1, 0.f) * ep;
        y2 = fmaxf(y2, 0.f) * ep; y3 = fmaxf(y3, 0.f) * ep;
        ((float4*)outf)[gr * 32 + lane] = make_float4(y0, y1, y2, y3);
        ((uint2*)outb)[gr * 32 + lane] = make_uint2(f2bf2(y0, y1), f2bf2(y2, y3));
      } else {
        cs0 += y0; cs1 += y1; cs2 += y2; cs3 += y3;
      }
    }
  }
  if constexpr (MODE == 3) {
    const float inv = 1.f / (float)NN;
    atomicAdd(&dout[c + 0], cs0 * inv);
    atomicAdd(&dout[c + 1], cs1 * inv);
    atomicAdd(&dout[c + 2], cs2 * inv);
    atomicAdd(&dout[c + 3], cs3 * inv);
  }
}

// ---------------- CSR gather + fused LayerNorm (one wave per target node) ----------------
// 4-wide unrolled: 4 independent 256B row reads in flight per wave, plus
// global_prefetch of the next group's rows to hide L2 latency.
__global__ __launch_bounds__(256) void k_gather(
    const u32* __restrict__ tmpu, const int* __restrict__ off, const int* __restrict__ srcA,
    const float* __restrict__ dis, const float* __restrict__ bias,
    const float* __restrict__ lng, const float* __restrict__ lnb,
    float* __restrict__ outf, u16* __restrict__ outb) {
  const int lane = threadIdx.x & 31;
  const int node = blockIdx.x * 8 + (threadIdx.x >> 5);
  if (node >= NN) return;
  const uint2* __restrict__ rows = (const uint2*)tmpu;
  float a0 = 0.f, a1 = 0.f, a2 = 0.f, a3 = 0.f;
  const int s = off[node], e = off[node + 1];
  int j = s;
  for (; j + 3 < e; j += 4) {
    const int s0 = srcA[j + 0], s1i = srcA[j + 1];
    const int s2i = srcA[j + 2], s3i = srcA[j + 3];
    if (j + 7 < e) {  // prefetch next group's rows (coalesced 256B per row)
      __builtin_prefetch(&rows[srcA[j + 4] * 32 + lane], 0, 3);
      __builtin_prefetch(&rows[srcA[j + 5] * 32 + lane], 0, 3);
      __builtin_prefetch(&rows[srcA[j + 6] * 32 + lane], 0, 3);
      __builtin_prefetch(&rows[srcA[j + 7] * 32 + lane], 0, 3);
    }
    const uint2 d0 = rows[s0 * 32 + lane];
    const uint2 d1 = rows[s1i * 32 + lane];
    const uint2 d2 = rows[s2i * 32 + lane];
    const uint2 d3 = rows[s3i * 32 + lane];
    a0 += bf2f(d0.x & 0xffffu) + bf2f(d1.x & 0xffffu) + bf2f(d2.x & 0xffffu) + bf2f(d3.x & 0xffffu);
    a1 += bf2f(d0.x >> 16)     + bf2f(d1.x >> 16)     + bf2f(d2.x >> 16)     + bf2f(d3.x >> 16);
    a2 += bf2f(d0.y & 0xffffu) + bf2f(d1.y & 0xffffu) + bf2f(d2.y & 0xffffu) + bf2f(d3.y & 0xffffu);
    a3 += bf2f(d0.y >> 16)     + bf2f(d1.y >> 16)     + bf2f(d2.y >> 16)     + bf2f(d3.y >> 16);
  }
  for (; j < e; ++j) {
    const uint2 d = rows[srcA[j] * 32 + lane];
    a0 += bf2f(d.x & 0xffffu); a1 += bf2f(d.x >> 16);
    a2 += bf2f(d.y & 0xffffu); a3 += bf2f(d.y >> 16);
  }
  const int c = lane * 4;
  const float dn = dis[node];
  float v0 = a0 * dn + bias[c + 0], v1 = a1 * dn + bias[c + 1];
  float v2 = a2 * dn + bias[c + 2], v3 = a3 * dn + bias[c + 3];
  float s1 = v0 + v1 + v2 + v3;
  float s2 = v0 * v0 + v1 * v1 + v2 * v2 + v3 * v3;
#pragma unroll
  for (int o = 16; o >= 1; o >>= 1) {
    s1 += __shfl_xor(s1, o, 32);
    s2 += __shfl_xor(s2, o, 32);
  }
  const float mu = s1 * (1.f / 128.f);
  const float rs = rsqrtf(s2 * (1.f / 128.f) - mu * mu + EPS_LN);
  const float y0 = (v0 - mu) * rs * lng[c + 0] + lnb[c + 0];
  const float y1 = (v1 - mu) * rs * lng[c + 1] + lnb[c + 1];
  const float y2 = (v2 - mu) * rs * lng[c + 2] + lnb[c + 2];
  const float y3 = (v3 - mu) * rs * lng[c + 3] + lnb[c + 3];
  ((float4*)outf)[node * 32 + lane] = make_float4(y0, y1, y2, y3);
  ((uint2*)outb)[node * 32 + lane] = make_uint2(f2bf2(y0, y1), f2bf2(y2, y3));
}

// ---------------- RK4 stage finisher: k = tanh(hc) + res_w*h0, stage update ----------------
__global__ void k_finish(const float* __restrict__ hc, const float* __restrict__ fin,
                         float* __restrict__ hh, float* __restrict__ ksum,
                         const float* __restrict__ resw, const float* __restrict__ scal,
                         float* __restrict__ outf, u16* __restrict__ outb, int s) {
  const int i = blockIdx.x * 256 + threadIdx.x;
  if (i >= NN * DH) return;
  const float dt = scal[0];
  const float k = tanhf(hc[i]) + resw[0] * fin[i];
  float un;
  if (s == 0)      { ksum[i] = k;          un = hh[i] + 0.5f * dt * k; }
  else if (s == 1) { ksum[i] += 2.f * k;   un = hh[i] + 0.5f * dt * k; }
  else if (s == 2) { ksum[i] += 2.f * k;   un = hh[i] + dt * k; }
  else             { un = hh[i] + (dt / 6.f) * (ksum[i] + k); }
  outf[i] = un;
  outb[i] = f2bf(un);
}

extern "C" void kernel_launch(void* const* d_in, const int* in_sizes, int n_in,
                              void* d_out, int out_size, void* d_ws, size_t ws_size,
                              hipStream_t stream) {
  (void)in_sizes; (void)n_in; (void)out_size; (void)ws_size;
  const float* x       = (const float*)d_in[0];
  const int*   ei      = (const int*)  d_in[1];
  const float* Wi      = (const float*)d_in[2];
  const float* bi      = (const float*)d_in[3];
  const float* ln_in_g = (const float*)d_in[4];
  const float* ln_in_b = (const float*)d_in[5];
  const float* meth    = (const float*)d_in[6];
  const float* hist    = (const float*)d_in[7];
  const float* gcn_w   = (const float*)d_in[8];
  const float* gcn_b   = (const float*)d_in[9];
  const float* ln_g    = (const float*)d_in[10];
  const float* ln_b    = (const float*)d_in[11];
  const float* gate_w  = (const float*)d_in[12];
  const float* gate_b  = (const float*)d_in[13];
  const float* res_w   = (const float*)d_in[14];
  const float* log_d   = (const float*)d_in[15];
  const float* Wo      = (const float*)d_in[16];
  const float* bo      = (const float*)d_in[17];
  const float* ln_o_g  = (const float*)d_in[18];
  const float* ln_o_b  = (const float*)d_in[19];
  float* dout = (float*)d_out;

  char* wp = (char*)d_ws;
  auto alloc = [&](size_t bytes) -> void* {
    void* p = (void*)wp;
    wp += (bytes + 255) & ~(size_t)255;
    return p;
  };
  float* scal = (float*)alloc(256);
  float* deg  = (float*)alloc((size_t)NN * 4);
  float* dis  = (float*)alloc((size_t)NN * 4);
  int*   off  = (int*)  alloc((size_t)(NN + 1) * 4);
  int*   cur  = (int*)  alloc((size_t)NN * 4);
  int*   srcA = (int*)  alloc((size_t)EE * 4);
  u16*   xb   = (u16*)  alloc((size_t)NN * 64 * 2);
  u16*   wti  = (u16*)  alloc((size_t)64 * 128 * 2);
  u16*   wtg  = (u16*)  alloc((size_t)3 * 128 * 128 * 2);
  u16*   wtgt = (u16*)  alloc((size_t)256 * 128 * 2);
  u16*   wto  = (u16*)  alloc((size_t)128 * 128 * 2);
  float* hhf  = (float*)alloc((size_t)NN * DH * 4);
  u16*   hhb  = (u16*)  alloc((size_t)NN * DH * 2);
  float* uf   = (float*)alloc((size_t)NN * DH * 4);
  u16*   ub   = (u16*)  alloc((size_t)NN * DH * 2);
  float* ks   = (float*)alloc((size_t)NN * DH * 4);
  u16*   tmpb = (u16*)  alloc((size_t)NN * DH * 2);
  float* bAf  = (float*)alloc((size_t)NN * DH * 4);
  u16*   bAb  = (u16*)  alloc((size_t)NN * DH * 2);
  float* bBf  = (float*)alloc((size_t)NN * DH * 4);
  u16*   bBb  = (u16*)  alloc((size_t)NN * DH * 2);

  hipMemsetAsync(deg, 0, (size_t)NN * 4, stream);
  hipMemsetAsync(cur, 0, (size_t)NN * 4, stream);

  k_prep<<<1, 128, 0, stream>>>(meth, hist, log_d, scal, dout);
  k_deg<<<EE / 256, 256, 0, stream>>>(ei, deg);
  k_dis<<<(NN + 255) / 256, 256, 0, stream>>>(deg, dis);
  k_scan<<<1, 1024, 0, stream>>>(deg, off);
  k_fill<<<EE / 256, 256, 0, stream>>>(ei, off, cur, srcA);

  k_conv<<<(NN * 64 + 255) / 256, 256, 0, stream>>>(x, xb, NN * 64);
  k_convT<<<(64 * 128 + 255) / 256, 256, 0, stream>>>(Wi, wti, 64, 128);
  k_convT<<<(128 * 128 + 255) / 256, 256, 0, stream>>>(gcn_w, wtg, 128, 128);
  k_convT<<<(128 * 128 + 255) / 256, 256, 0, stream>>>(gcn_w + 128 * 128, wtg + 128 * 128, 128, 128);
  k_convT<<<(128 * 128 + 255) / 256, 256, 0, stream>>>(gcn_w + 2 * 128 * 128, wtg + 2 * 128 * 128, 128, 128);
  k_convT<<<(256 * 128 + 255) / 256, 256, 0, stream>>>(gate_w, wtgt, 256, 128);
  k_convT<<<(128 * 128 + 255) / 256, 256, 0, stream>>>(Wo, wto, 128, 128);

  const int GB = (NN + 63) / 64;  // GEMM grid: 64 rows / block
  const int GG = (NN + 7) / 8;    // gather grid: 8 nodes / block

  // input projection -> hh (f32 + bf16)
  k_gemm<1, 32, 0><<<GB, 128, 0, stream>>>(xb, nullptr, wti, bi, ln_in_g, ln_in_b,
                                           nullptr, nullptr, hhf, hhb, scal, nullptr);

  for (int st = 0; st < TP - 1; ++st) {
    for (int s = 0; s < 4; ++s) {
      const float* finf = (s == 0) ? hhf : uf;
      const u16*   finb = (s == 0) ? hhb : ub;
      // GCN layer 0
      k_gemm<1, 64, 1><<<GB, 128, 0, stream>>>(finb, nullptr, wtg, nullptr, nullptr, nullptr,
                                               dis, nullptr, nullptr, tmpb, nullptr, nullptr);
      k_gather<<<GG, 256, 0, stream>>>((const u32*)tmpb, off, srcA, dis,
                                       gcn_b, ln_g, ln_b, bAf, bAb);
      // GCN layer 1 + gate (hc = bufA, hn = bufB -> result in bufB)
      k_gemm<1, 64, 1><<<GB, 128, 0, stream>>>(bAb, nullptr, wtg + 128 * 128, nullptr, nullptr,
                                               nullptr, dis, nullptr, nullptr, tmpb, nullptr, nullptr);
      k_gather<<<GG, 256, 0, stream>>>((const u32*)tmpb, off, srcA, dis,
                                       gcn_b + 128, ln_g + 128, ln_b + 128, bBf, bBb);
      k_gemm<2, 64, 2><<<GB, 128, 0, stream>>>(bAb, bBb, wtgt, gate_b, nullptr, nullptr,
                                               nullptr, bAf, bBf, bBb, nullptr, nullptr);
      // GCN layer 2 + gate (hc = bufB, hn = bufA -> result in bufA)
      k_gemm<1, 64, 1><<<GB, 128, 0, stream>>>(bBb, nullptr, wtg + 2 * 128 * 128, nullptr, nullptr,
                                               nullptr, dis, nullptr, nullptr, tmpb, nullptr, nullptr);
      k_gather<<<GG, 256, 0, stream>>>((const u32*)tmpb, off, srcA, dis,
                                       gcn_b + 256, ln_g + 256, ln_b + 256, bAf, bAb);
      k_gemm<2, 64, 2><<<GB, 128, 0, stream>>>(bBb, bAb, wtgt, gate_b, nullptr, nullptr,
                                               nullptr, bBf, bAf, bAb, nullptr, nullptr);
      // RK4 stage update
      float* of = (s < 3) ? uf : hhf;
      u16*   ob = (s < 3) ? ub : hhb;
      k_finish<<<(NN * DH + 255) / 256, 256, 0, stream>>>(bAf, finf, hhf, ks, res_w, scal,
                                                          of, ob, s);
    }
  }
  // output projection + LN + global mean pool
  k_gemm<1, 64, 3><<<GB, 128, 0, stream>>>(hhb, nullptr, wto, bo, ln_o_g, ln_o_b,
                                           nullptr, nullptr, nullptr, nullptr, scal, dout);
}
